// GNN_9457517986237
// MI455X (gfx1250) — compile-verified, matrix operand order checked by reading, not run on hardware
//
#include <hip/hip_runtime.h>

typedef __attribute__((ext_vector_type(16))) __bf16 v16bf;
typedef __attribute__((ext_vector_type(8)))  float  v8f;

#define HID 64
#define EA_DIM 16
#define LDT 68  // padded LDS tile row stride (floats) to spread banks

// ---------- WMMA fragment helpers (bf16 A/B, f32 C) ----------
// A 16x32 bf16 layout: lane L -> row M=L&15; sel=L>>4;
//   halves 0..7  -> K = kwin + sel*8 + h
//   halves 8..15 -> K = kwin + sel*8 + 16 + h
__device__ __forceinline__ v16bf make_a_frag(const float* __restrict__ row, int kwin, int sel) {
    v16bf a;
    const float* p = row + kwin + sel * 8;
#pragma unroll
    for (int h = 0; h < 8; ++h) {
        a[h]     = (__bf16)p[h];
        a[h + 8] = (__bf16)p[16 + h];
    }
    return a;
}

// A for K=16 matrix padded to K=32 (upper halves zero)
__device__ __forceinline__ v16bf make_a_frag_k16(const float* __restrict__ row, int sel) {
    v16bf a;
    const float* p = row + sel * 8;
#pragma unroll
    for (int h = 0; h < 8; ++h) {
        a[h]     = (__bf16)p[h];
        a[h + 8] = (__bf16)0.0f;
    }
    return a;
}

// B 32x16 bf16 layout (mirror of A indexed by N): lane L -> col n; same K pattern.
__device__ __forceinline__ v16bf make_b_frag(const float* __restrict__ W, int K, int n, int kwin, int sel) {
    v16bf b;
    const int kb = kwin + sel * 8;
#pragma unroll
    for (int h = 0; h < 8; ++h) {
        const int ka = kb + h;
        const int kc = kb + 16 + h;
        b[h]     = (ka < K) ? (__bf16)W[ka * 64 + n] : (__bf16)0.0f;
        b[h + 8] = (kc < K) ? (__bf16)W[kc * 64 + n] : (__bf16)0.0f;
    }
    return b;
}

// C/D f32 16x16: lane L -> N=L&15 (+nt*16); vgpr j -> M = j + 8*(L>>4)
__device__ __forceinline__ void store_c_tile(float* __restrict__ tile, const v8f& c,
                                             int nt, int r, int sel, const float* __restrict__ bias) {
    const int n  = nt * 16 + r;
    const float bv = bias[n];
#pragma unroll
    for (int j = 0; j < 8; ++j) tile[(j + 8 * sel) * LDT + n] = c[j] + bv;
}

// Row-wise LayerNorm + ReLU over 64 features; lanes 0..15 each own one row.
__device__ __forceinline__ void ln_relu_rows(float* __restrict__ tile, const float* __restrict__ g,
                                             const float* __restrict__ be, int lane) {
    if (lane < 16) {
        float* row = tile + lane * LDT;
        float s = 0.f;
#pragma unroll 8
        for (int c = 0; c < 64; ++c) s += row[c];
        const float mean = s * (1.0f / 64.0f);
        float v = 0.f;
#pragma unroll 8
        for (int c = 0; c < 64; ++c) { const float d = row[c] - mean; v += d * d; }
        const float inv = rsqrtf(v * (1.0f / 64.0f) + 1e-5f);
#pragma unroll 8
        for (int c = 0; c < 64; ++c) {
            const float y = (row[c] - mean) * inv * g[c] + be[c];
            row[c] = fmaxf(y, 0.f);
        }
    }
}

__device__ __forceinline__ float bf16_bits_to_f32(unsigned v) {
    union { unsigned u; float f; } c;
    c.u = v << 16;
    return c.f;
}

// ---------- Kernel 1: fused edge-attr MLP ([E,16] -> bf16 ea[E,64]) ----------
// Input tiles (16 edges x 16 attrs = 1KB fp32) are staged into LDS with the
// gfx1250 async-to-LDS DMA path (ASYNCcnt), then consumed as WMMA A fragments.
__global__ __launch_bounds__(256) void ea_mlp_kernel(
    const float* __restrict__ edge_attr,
    const float* __restrict__ W1, const float* __restrict__ b1,
    const float* __restrict__ g1, const float* __restrict__ be1,
    const float* __restrict__ W2, const float* __restrict__ b2,
    const float* __restrict__ g2, const float* __restrict__ be2,
    __bf16* __restrict__ ea_out, int numTiles)
{
    __shared__ float tiles[8][16 * LDT];
    __shared__ float stage[8][16 * EA_DIM];   // 1KB per wave async staging
    const int lane = threadIdx.x & 31;
    const int wave = threadIdx.x >> 5;
    const int r    = lane & 15;
    const int sel  = lane >> 4;
    float* tile   = &tiles[wave][0];
    float* stagew = &stage[wave][0];
    // LDS byte offset of this lane's 16B chunk in the staging buffer
    // (generic shared address truncates to LDS offset: addr[31:0])
    const unsigned ldsa = (unsigned)(uintptr_t)stagew + (unsigned)lane * 16u;

    // Loop-invariant B fragments
    v16bf B1[4];
#pragma unroll
    for (int nt = 0; nt < 4; ++nt) B1[nt] = make_b_frag(W1, EA_DIM, nt * 16 + r, 0, sel);
    v16bf B2[4][2];
#pragma unroll
    for (int nt = 0; nt < 4; ++nt)
#pragma unroll
        for (int ks = 0; ks < 2; ++ks) B2[nt][ks] = make_b_frag(W2, HID, nt * 16 + r, ks * 32, sel);

    for (int t = blockIdx.x * 8 + wave; t < numTiles; t += gridDim.x * 8) {
        // --- async DMA: 1KB tile of edge_attr -> LDS staging (2 x b128/lane) ---
        const unsigned long long ga =
            (unsigned long long)(uintptr_t)edge_attr + (unsigned long long)t * (16 * EA_DIM * 4) + (unsigned long long)lane * 16u;
        asm volatile("global_load_async_to_lds_b128 %0, %1, off"
                     :: "v"(ldsa), "v"(ga) : "memory");
        asm volatile("global_load_async_to_lds_b128 %0, %1, off offset:512"
                     :: "v"(ldsa), "v"(ga) : "memory");
        __builtin_prefetch(edge_attr + (size_t)(t + gridDim.x * 8) * 16 * EA_DIM, 0, 1);
        asm volatile("s_wait_asynccnt 0x0" ::: "memory");

        // GEMM1: [16x16(pad32)] x [16x64], A from staged LDS tile
        const v16bf a1 = make_a_frag_k16(stagew + r * EA_DIM, sel);
#pragma unroll
        for (int nt = 0; nt < 4; ++nt) {
            v8f c = {0.f, 0.f, 0.f, 0.f, 0.f, 0.f, 0.f, 0.f};
            c = __builtin_amdgcn_wmma_f32_16x16x32_bf16(false, a1, false, B1[nt], (short)0, c, false, false);
            store_c_tile(tile, c, nt, r, sel, b1);
        }
        ln_relu_rows(tile, g1, be1, lane);

        // GEMM2: [16x64] x [64x64], A from the LN'd tile in LDS
        v8f c2[4];
#pragma unroll
        for (int nt = 0; nt < 4; ++nt) c2[nt] = (v8f){0.f, 0.f, 0.f, 0.f, 0.f, 0.f, 0.f, 0.f};
#pragma unroll
        for (int ks = 0; ks < 2; ++ks) {
            const v16bf a2 = make_a_frag(tile + r * LDT, ks * 32, sel);
#pragma unroll
            for (int nt = 0; nt < 4; ++nt)
                c2[nt] = __builtin_amdgcn_wmma_f32_16x16x32_bf16(false, a2, false, B2[nt][ks], (short)0, c2[nt], false, false);
        }
#pragma unroll
        for (int nt = 0; nt < 4; ++nt) store_c_tile(tile, c2[nt], nt, r, sel, b2);
        ln_relu_rows(tile, g2, be2, lane);

        // Emit bf16 ea (halves the dominant HBM stream; re-read 3x)
        __bf16* outp = ea_out + (size_t)t * 16 * HID;
        for (int idx = lane; idx < 16 * 64; idx += 32) {
            const int m = idx >> 6, cc = idx & 63;
            outp[idx] = (__bf16)tile[m * LDT + cc];
        }
    }
}

// ---------- Kernel 2: node MLP  h = relu(LN(x@W + b)); also seeds xnext = h ----------
__global__ __launch_bounds__(256) void node_mlp_kernel(
    const float* __restrict__ x,
    const float* __restrict__ W, const float* __restrict__ b,
    const float* __restrict__ g, const float* __restrict__ be,
    float* __restrict__ h, float* __restrict__ xnext, int numTiles)
{
    __shared__ float tiles[8][16 * LDT];
    const int lane = threadIdx.x & 31;
    const int wave = threadIdx.x >> 5;
    const int r    = lane & 15;
    const int sel  = lane >> 4;
    float* tile = &tiles[wave][0];

    v16bf B[4][2];
#pragma unroll
    for (int nt = 0; nt < 4; ++nt)
#pragma unroll
        for (int ks = 0; ks < 2; ++ks) B[nt][ks] = make_b_frag(W, HID, nt * 16 + r, ks * 32, sel);

    for (int t = blockIdx.x * 8 + wave; t < numTiles; t += gridDim.x * 8) {
        const float* Arow = x + (size_t)(t * 16 + r) * HID;
        v8f c[4];
#pragma unroll
        for (int nt = 0; nt < 4; ++nt) c[nt] = (v8f){0.f, 0.f, 0.f, 0.f, 0.f, 0.f, 0.f, 0.f};
#pragma unroll
        for (int ks = 0; ks < 2; ++ks) {
            const v16bf a = make_a_frag(Arow, ks * 32, sel);
#pragma unroll
            for (int nt = 0; nt < 4; ++nt)
                c[nt] = __builtin_amdgcn_wmma_f32_16x16x32_bf16(false, a, false, B[nt][ks], (short)0, c[nt], false, false);
        }
#pragma unroll
        for (int nt = 0; nt < 4; ++nt) store_c_tile(tile, c[nt], nt, r, sel, b);
        ln_relu_rows(tile, g, be, lane);

        float* hp = h     + (size_t)t * 16 * HID;
        float* xp = xnext + (size_t)t * 16 * HID;
        for (int idx = lane; idx < 16 * 64; idx += 32) {
            const int m = idx >> 6, cc = idx & 63;
            const float v = tile[m * LDT + cc];
            hp[idx] = v;
            xp[idx] = v;
        }
    }
}

// ---------- Kernel 3: edge message  xnext[src] += h[dst] * ea  (16 thr/edge, float4) ----------
__global__ __launch_bounds__(256) void edge_msg_kernel(
    const float* __restrict__ h, const __bf16* __restrict__ ea,
    const int* __restrict__ edge_index, float* __restrict__ xnext, int E)
{
    const int gid = blockIdx.x * blockDim.x + threadIdx.x;
    const int e = gid >> 4;
    if (e >= E) return;
    const int f = (gid & 15) * 4;
    const int src = edge_index[e];
    const int dst = edge_index[E + e];
    const float4 hv = *(const float4*)(h + (size_t)dst * HID + f);
    // single b64 load on the dominant HBM stream (4 x bf16)
    const uint2 epk = *(const uint2*)(ea + (size_t)e * HID + f);
    const float e0 = bf16_bits_to_f32(epk.x & 0xffffu);
    const float e1 = bf16_bits_to_f32(epk.x >> 16);
    const float e2 = bf16_bits_to_f32(epk.y & 0xffffu);
    const float e3 = bf16_bits_to_f32(epk.y >> 16);
    float* xp = xnext + (size_t)src * HID + f;
    atomicAdd(xp + 0, hv.x * e0);
    atomicAdd(xp + 1, hv.y * e1);
    atomicAdd(xp + 2, hv.z * e2);
    atomicAdd(xp + 3, hv.w * e3);
}

// ---------- Kernel 4/5: pooling ----------
__global__ __launch_bounds__(256) void zero_kernel(float* __restrict__ p, int n) {
    const int i = blockIdx.x * blockDim.x + threadIdx.x;
    if (i < n) p[i] = 0.f;
}

__global__ __launch_bounds__(256) void pool_kernel(
    const float* __restrict__ x, const int* __restrict__ batch,
    float* __restrict__ pooled, int N)
{
    const int gid = blockIdx.x * blockDim.x + threadIdx.x;
    const int n = gid >> 4;
    if (n >= N) return;
    const int f = (gid & 15) * 4;
    const int gidx = batch[n];
    const float4 v = *(const float4*)(x + (size_t)n * HID + f);
    float* pp = pooled + (size_t)gidx * HID + f;
    atomicAdd(pp + 0, v.x);
    atomicAdd(pp + 1, v.y);
    atomicAdd(pp + 2, v.z);
    atomicAdd(pp + 3, v.w);
}

// ---------- Kernel 6: final GEMM  out = pooled @ Wo + bo ----------
__global__ __launch_bounds__(256) void out_gemm_kernel(
    const float* __restrict__ pooled, const float* __restrict__ Wo,
    const float* __restrict__ bo, float* __restrict__ out, int numTiles)
{
    const int lane = threadIdx.x & 31;
    const int wave = threadIdx.x >> 5;
    const int r    = lane & 15;
    const int sel  = lane >> 4;

    v16bf B[4][2];
#pragma unroll
    for (int nt = 0; nt < 4; ++nt)
#pragma unroll
        for (int ks = 0; ks < 2; ++ks) B[nt][ks] = make_b_frag(Wo, HID, nt * 16 + r, ks * 32, sel);

    for (int t = blockIdx.x * 8 + wave; t < numTiles; t += gridDim.x * 8) {
        const float* Arow = pooled + (size_t)(t * 16 + r) * HID;
        v8f c[4];
#pragma unroll
        for (int nt = 0; nt < 4; ++nt) c[nt] = (v8f){0.f, 0.f, 0.f, 0.f, 0.f, 0.f, 0.f, 0.f};
#pragma unroll
        for (int ks = 0; ks < 2; ++ks) {
            const v16bf a = make_a_frag(Arow, ks * 32, sel);
#pragma unroll
            for (int nt = 0; nt < 4; ++nt)
                c[nt] = __builtin_amdgcn_wmma_f32_16x16x32_bf16(false, a, false, B[nt][ks], (short)0, c[nt], false, false);
        }
#pragma unroll
        for (int nt = 0; nt < 4; ++nt) {
            const int n = nt * 16 + r;
            const float bv = bo[n];
#pragma unroll
            for (int j = 0; j < 8; ++j)
                out[(size_t)(t * 16 + j + 8 * sel) * HID + n] = c[nt][j] + bv;
        }
    }
}

extern "C" void kernel_launch(void* const* d_in, const int* in_sizes, int n_in,
                              void* d_out, int out_size, void* d_ws, size_t ws_size,
                              hipStream_t stream) {
    const float* x     = (const float*)d_in[0];
    const int*   ei    = (const int*)d_in[1];
    const float* eattr = (const float*)d_in[2];
    const int*   batch = (const int*)d_in[3];
    const float* eaW1  = (const float*)d_in[4];
    const float* eab1  = (const float*)d_in[5];
    const float* eag1  = (const float*)d_in[6];
    const float* eabe1 = (const float*)d_in[7];
    const float* eaW2  = (const float*)d_in[8];
    const float* eab2  = (const float*)d_in[9];
    const float* eag2  = (const float*)d_in[10];
    const float* eabe2 = (const float*)d_in[11];
    const float* Wo    = (const float*)d_in[24];
    const float* bo    = (const float*)d_in[25];

    const int N  = in_sizes[0] / HID;
    const int E  = in_sizes[1] / 2;
    const int Gn = out_size / HID;

    // Workspace carving (256B aligned)
    char* ws = (char*)d_ws;
    size_t off = 0;
    auto carve = [&](size_t bytes) -> void* {
        void* p = ws + off;
        off = (off + bytes + 255) & ~(size_t)255;
        return p;
    };
    __bf16* ea    = (__bf16*)carve((size_t)E * HID * sizeof(__bf16));
    float* bufA   = (float*)carve((size_t)N * HID * sizeof(float));
    float* bufB   = (float*)carve((size_t)N * HID * sizeof(float));
    float* hbuf   = (float*)carve((size_t)N * HID * sizeof(float));
    float* pooled = (float*)carve((size_t)Gn * HID * sizeof(float));
    (void)ws_size; (void)n_in;

    // 1) Edge-attr MLP (fused GEMM/LN/ReLU x2), async-LDS staged input, bf16 output
    const int tilesE = E / 16;
    ea_mlp_kernel<<<(tilesE + 7) / 8, 256, 0, stream>>>(
        eattr, eaW1, eab1, eag1, eabe1, eaW2, eab2, eag2, eabe2, ea, tilesE);

    // 2) Three message-passing layers
    const int tilesN = N / 16;
    const float* xin = x;
    float* bufs[2] = {bufA, bufB};
    for (int l = 0; l < 3; ++l) {
        const float* W  = (const float*)d_in[12 + 4 * l];
        const float* b  = (const float*)d_in[13 + 4 * l];
        const float* g  = (const float*)d_in[14 + 4 * l];
        const float* be = (const float*)d_in[15 + 4 * l];
        float* xnext = bufs[l & 1];
        node_mlp_kernel<<<(tilesN + 7) / 8, 256, 0, stream>>>(xin, W, b, g, be, hbuf, xnext, tilesN);
        const long edgeThreads = (long)E * 16;
        edge_msg_kernel<<<(int)((edgeThreads + 255) / 256), 256, 0, stream>>>(hbuf, ea, ei, xnext, E);
        xin = xnext;
    }

    // 3) Graph pooling + output GEMM
    const int npool = Gn * HID;
    zero_kernel<<<(npool + 255) / 256, 256, 0, stream>>>(pooled, npool);
    const long poolThreads = (long)N * 16;
    pool_kernel<<<(int)((poolThreads + 255) / 256), 256, 0, stream>>>(xin, batch, pooled, N);
    const int tilesG = Gn / 16;
    out_gemm_kernel<<<(tilesG + 7) / 8, 256, 0, stream>>>(pooled, Wo, bo, (float*)d_out, tilesG);
}
